// DGNLayer_40776419508435
// MI455X (gfx1250) — matrix operand order repaired and checked by reading, not compiled, not run on hardware
//
#include <hip/hip_runtime.h>

#define D 64
#define APAD 8   // 16-byte row padding to rotate LDS banks
#define FLT_MAX_C 3.402823466e38f

typedef __attribute__((ext_vector_type(16))) __bf16 bf16x16;
typedef __attribute__((ext_vector_type(8)))  __bf16 bf16x8;
typedef __attribute__((ext_vector_type(8)))  float  floatx8;

#define CAT16(lo, hi) __builtin_shufflevector(lo, hi, 0, 1, 2, 3, 4, 5, 6, 7, \
                                              8, 9, 10, 11, 12, 13, 14, 15)

__device__ __forceinline__ bf16x8 pack8(float4 v0, float4 v1) {
  bf16x8 p;
  p[0] = (__bf16)v0.x; p[1] = (__bf16)v0.y; p[2] = (__bf16)v0.z; p[3] = (__bf16)v0.w;
  p[4] = (__bf16)v1.x; p[5] = (__bf16)v1.y; p[6] = (__bf16)v1.z; p[7] = (__bf16)v1.w;
  return p;
}

// scalar |x| via inline asm: opaque to InstCombine/SLP, avoids the gfx1250 ISel
// crash on SLP-vectorized "v2f32 bitcast(and(bitcast))" fabs idiom
__device__ __forceinline__ float absf_dev(float x) {
  float r;
  asm("v_and_b32 %0, 0x7fffffff, %1" : "=v"(r) : "v"(x));
  return r;
}

__device__ __forceinline__ void atomicMaxF(float* addr, float val) {
  // dual-op trick: valid for mixed signs when memory is float-encoded
  if (val >= 0.0f) atomicMax((int*)addr, __float_as_int(val));
  else             atomicMin((unsigned int*)addr, __float_as_uint(val));
}

// ---------------------------------------------------------------- init
__global__ void dgn_init_kernel(float* __restrict__ agg_sum, float* __restrict__ agg_max,
                                float* __restrict__ agg_ew, float* __restrict__ deg,
                                float* __restrict__ absum, float* __restrict__ wsum,
                                float* __restrict__ stats, int N) {
  const size_t total = (size_t)N * D;
  for (size_t i = (size_t)blockIdx.x * blockDim.x + threadIdx.x; i < total;
       i += (size_t)gridDim.x * blockDim.x) {
    agg_sum[i] = 0.0f;
    agg_ew[i]  = 0.0f;
    agg_max[i] = -FLT_MAX_C;
    if (i < (size_t)N) { deg[i] = 0.0f; absum[i] = 0.0f; wsum[i] = 0.0f; }
    if (i < 2 * D) stats[i] = 0.0f;
  }
}

// ---------------------------------------------------------------- pre: deg, absum
__global__ void dgn_pre_kernel(const float* __restrict__ eig, const int* __restrict__ dst,
                               float* __restrict__ deg, float* __restrict__ absum, int E) {
  const int i = blockIdx.x * blockDim.x + threadIdx.x;
  if (i >= E) return;
  const int d = dst[i];
  const float a = absf_dev(eig[i]);
  atomicAdd(&deg[d], 1.0f);
  atomicAdd(&absum[d], a);
}

// ---------------------------------------------------------------- edge GEMM + scatter
// per wave: 16 edges x 64 outputs, K=128 via 4x wmma_f32_16x16x32_bf16 per n-tile
__global__ __launch_bounds__(256) void dgn_edge_kernel(
    const float* __restrict__ h, const float* __restrict__ eig,
    const int* __restrict__ src, const int* __restrict__ dst,
    const float* __restrict__ W1, const float* __restrict__ b1,
    const float* __restrict__ absum,
    float* __restrict__ agg_sum, float* __restrict__ agg_max,
    float* __restrict__ agg_ew, float* __restrict__ wsum, int E) {
  __shared__ __bf16 Bs[D][128 + APAD];      // Bs[o][k] = W1[o][k]  (K-contiguous per out)
  __shared__ __bf16 As[8][16][128 + APAD];  // per-wave z2 tiles    (K-contiguous per row)

  const int tid  = threadIdx.x;
  const int wave = tid >> 5;
  const int lane = tid & 31;

  // stage W1 (row-major already matches B-fragment layout), 8 bf16 per chunk
  for (int c = tid; c < (D * 128) / 8; c += 256) {
    const int o  = c >> 4;          // 16 chunks per 128-wide row
    const int k8 = (c & 15) * 8;
    const float4* wp = (const float4*)(W1 + o * 128 + k8);
    *(bf16x8*)&Bs[o][k8] = pack8(wp[0], wp[1]);
  }

  const int eBase = blockIdx.x * 128 + wave * 16;
  {
    const int row  = lane >> 1;
    const int side = lane & 1;            // 0 -> h[src], 1 -> h[dst]
    const int eIdx = eBase + row;
    __bf16* arow = &As[wave][row][side * D];
    if (eIdx < E) {
      const int node = side ? dst[eIdx] : src[eIdx];
      const float4* hr = (const float4*)(h + (size_t)node * D);
#pragma unroll
      for (int g = 0; g < 8; ++g)
        *(bf16x8*)&arow[g * 8] = pack8(hr[2 * g], hr[2 * g + 1]);
    } else {
      const bf16x8 z = {};
#pragma unroll
      for (int g = 0; g < 8; ++g) *(bf16x8*)&arow[g * 8] = z;
    }
  }
  __syncthreads();

  const int m    = lane & 15;
  const int half = lane >> 4;

  floatx8 acc[4] = {};

  const __bf16* amrow = As[wave][m];
#pragma unroll
  for (int kc = 0; kc < 4; ++kc) {
    // A-frag: two contiguous 8-elem runs at k = kc*32 + half*8 (+16)
    const bf16x8 a0 = *(const bf16x8*)&amrow[kc * 32 + half * 8];
    const bf16x8 a1 = *(const bf16x8*)&amrow[kc * 32 + half * 8 + 16];
    const bf16x16 afrag = CAT16(a0, a1);
#pragma unroll
    for (int nt = 0; nt < 4; ++nt) {
      // B-frag: two contiguous 8-elem runs at k = kc*32 + half*8 (+16)
      const __bf16* brow = Bs[nt * 16 + m];
      const int k0 = kc * 32 + half * 8;
      const bf16x8 b0 = *(const bf16x8*)&brow[k0];
      const bf16x8 b1 = *(const bf16x8*)&brow[k0 + 16];
      const bf16x16 bfrag = CAT16(b0, b1);
      acc[nt] = __builtin_amdgcn_wmma_f32_16x16x32_bf16(
          false, afrag, false, bfrag, (short)0, acc[nt], false, false);
    }
  }

  // epilogue: D/C layout row M = r + 8*half, col = nt*16 + m
#pragma unroll
  for (int r = 0; r < 8; ++r) {
    const int eIdx = eBase + r + 8 * half;
    if (eIdx >= E) continue;
    const int d = dst[eIdx];
    const float wv = eig[eIdx] / (absum[d] + 1e-8f);
    if (m == 0) atomicAdd(&wsum[d], wv);
#pragma unroll
    for (int nt = 0; nt < 4; ++nt) {
      const int col = nt * 16 + m;
      const float ev = acc[nt][r] + b1[col];
      const size_t o = (size_t)d * D + col;
      atomicAdd(&agg_sum[o], ev);
      atomicMaxF(&agg_max[o], ev);
      atomicAdd(&agg_ew[o], ev * wv);
    }
  }
}

// ---------------------------------------------------------------- node GEMM + BN stats
// per wave: 16 nodes x 64 outputs, K=256 in two 128-col phases
__global__ __launch_bounds__(128) void dgn_node_kernel(
    const float* __restrict__ h, const float* __restrict__ snorm,
    const float* __restrict__ W2, const float* __restrict__ b2,
    const float* __restrict__ agg_sum, const float* __restrict__ agg_max,
    const float* __restrict__ agg_ew, const float* __restrict__ deg,
    const float* __restrict__ wsum,
    float* __restrict__ h2, float* __restrict__ stats, int N) {
  __shared__ __bf16 Bs[D][256 + APAD];      // Bs[o][k] = W2[o][k]
  __shared__ __bf16 As[4][16][128 + APAD];  // per-wave hc half-tiles
  __shared__ float redsum[D];
  __shared__ float redsq[D];

  const int tid  = threadIdx.x;
  const int wave = tid >> 5;
  const int lane = tid & 31;

  for (int c = tid; c < (D * 256) / 8; c += 128) {
    const int o  = c >> 5;          // 32 chunks per 256-wide row
    const int k8 = (c & 31) * 8;
    const float4* wp = (const float4*)(W2 + o * 256 + k8);
    *(bf16x8*)&Bs[o][k8] = pack8(wp[0], wp[1]);
  }
  if (tid < D) { redsum[tid] = 0.0f; redsq[tid] = 0.0f; }

  const int nBase = blockIdx.x * 64 + wave * 16;
  const int row   = lane >> 1;
  const int chal  = lane & 1;
  const int node  = nBase + row;
  const int m     = lane & 15;
  const int half  = lane >> 4;

  floatx8 acc[4] = {};

  for (int ph = 0; ph < 2; ++ph) {
    __syncthreads();  // As from previous phase fully consumed
    __bf16* arow = &As[wave][row][chal * D];
    const int sel = ph * 2 + chal;  // 0:h 1:mean 2:max 3:dir
    if (node < N) {
      const size_t base = (size_t)node * D;
      if (sel == 0) {
        const float4* hp = (const float4*)(h + base);
#pragma unroll
        for (int g = 0; g < 8; ++g)
          *(bf16x8*)&arow[g * 8] = pack8(hp[2 * g], hp[2 * g + 1]);
      } else if (sel == 1) {
        const float inv = 1.0f / fmaxf(deg[node], 1.0f);
        const float4* sp = (const float4*)(agg_sum + base);
#pragma unroll
        for (int g = 0; g < 8; ++g) {
          float4 v0 = sp[2 * g], v1 = sp[2 * g + 1];
          v0.x *= inv; v0.y *= inv; v0.z *= inv; v0.w *= inv;
          v1.x *= inv; v1.y *= inv; v1.z *= inv; v1.w *= inv;
          *(bf16x8*)&arow[g * 8] = pack8(v0, v1);
        }
      } else if (sel == 2) {
        const float sc = (deg[node] > 0.0f) ? 1.0f : 0.0f;
        const float4* mp = (const float4*)(agg_max + base);
#pragma unroll
        for (int g = 0; g < 8; ++g) {
          float4 v0 = mp[2 * g], v1 = mp[2 * g + 1];
          v0.x *= sc; v0.y *= sc; v0.z *= sc; v0.w *= sc;
          v1.x *= sc; v1.y *= sc; v1.z *= sc; v1.w *= sc;
          *(bf16x8*)&arow[g * 8] = pack8(v0, v1);
        }
      } else {
        const float ws = wsum[node];
        const float4* ep = (const float4*)(agg_ew + base);
        const float4* hp = (const float4*)(h + base);
#pragma unroll
        for (int g = 0; g < 8; ++g) {
          float4 e0 = ep[2 * g], e1 = ep[2 * g + 1];
          float4 h0 = hp[2 * g], h1 = hp[2 * g + 1];
          e0.x = absf_dev(e0.x - ws * h0.x); e0.y = absf_dev(e0.y - ws * h0.y);
          e0.z = absf_dev(e0.z - ws * h0.z); e0.w = absf_dev(e0.w - ws * h0.w);
          e1.x = absf_dev(e1.x - ws * h1.x); e1.y = absf_dev(e1.y - ws * h1.y);
          e1.z = absf_dev(e1.z - ws * h1.z); e1.w = absf_dev(e1.w - ws * h1.w);
          *(bf16x8*)&arow[g * 8] = pack8(e0, e1);
        }
      }
    } else {
      const bf16x8 z = {};
#pragma unroll
      for (int g = 0; g < 8; ++g) *(bf16x8*)&arow[g * 8] = z;
    }
    __syncthreads();

    const __bf16* amrow = As[wave][m];
#pragma unroll
    for (int kc = 0; kc < 4; ++kc) {
      const bf16x8 a0 = *(const bf16x8*)&amrow[kc * 32 + half * 8];
      const bf16x8 a1 = *(const bf16x8*)&amrow[kc * 32 + half * 8 + 16];
      const bf16x16 afrag = CAT16(a0, a1);
#pragma unroll
      for (int nt = 0; nt < 4; ++nt) {
        const __bf16* brow = Bs[nt * 16 + m];
        const int k0 = ph * 128 + kc * 32 + half * 8;
        const bf16x8 b0 = *(const bf16x8*)&brow[k0];
        const bf16x8 b1 = *(const bf16x8*)&brow[k0 + 16];
        const bf16x16 bfrag = CAT16(b0, b1);
        acc[nt] = __builtin_amdgcn_wmma_f32_16x16x32_bf16(
            false, afrag, false, bfrag, (short)0, acc[nt], false, false);
      }
    }
  }

  float lsum[4] = {0.f, 0.f, 0.f, 0.f}, lsq[4] = {0.f, 0.f, 0.f, 0.f};
#pragma unroll
  for (int r = 0; r < 8; ++r) {
    const int n = nBase + r + 8 * half;
    if (n >= N) continue;
    const float sn = snorm[n];
#pragma unroll
    for (int nt = 0; nt < 4; ++nt) {
      const int col = nt * 16 + m;
      const float v = (acc[nt][r] + b2[col]) * sn;
      h2[(size_t)n * D + col] = v;
      lsum[nt] += v;
      lsq[nt]  += v * v;
    }
  }
#pragma unroll
  for (int nt = 0; nt < 4; ++nt) {
    atomicAdd(&redsum[nt * 16 + m], lsum[nt]);
    atomicAdd(&redsq[nt * 16 + m],  lsq[nt]);
  }
  __syncthreads();
  if (tid < D) {
    atomicAdd(&stats[tid],     redsum[tid]);
    atomicAdd(&stats[D + tid], redsq[tid]);
  }
}

// ---------------------------------------------------------------- BN + relu + residual
__global__ void dgn_final_kernel(const float* __restrict__ h, const float* __restrict__ h2,
                                 const float* __restrict__ stats,
                                 const float* __restrict__ gamma, const float* __restrict__ beta,
                                 float* __restrict__ out, int N) {
  const int i = blockIdx.x * blockDim.x + threadIdx.x;
  if (i >= N * D) return;
  const int c = i & (D - 1);
  const float invN = 1.0f / (float)N;
  const float mu  = stats[c] * invN;
  const float var = stats[D + c] * invN - mu * mu;
  const float x = (h2[i] - mu) * rsqrtf(var + 1e-5f) * gamma[c] + beta[c];
  out[i] = h[i] + fmaxf(x, 0.0f);
}

// ---------------------------------------------------------------- launcher
extern "C" void kernel_launch(void* const* d_in, const int* in_sizes, int n_in,
                              void* d_out, int out_size, void* d_ws, size_t ws_size,
                              hipStream_t stream) {
  const float* h     = (const float*)d_in[0];
  const float* eig   = (const float*)d_in[1];
  const float* snorm = (const float*)d_in[2];
  const int*   src   = (const int*)d_in[3];
  const int*   dst   = (const int*)d_in[4];
  const float* W1    = (const float*)d_in[5];
  const float* b1    = (const float*)d_in[6];
  const float* W2    = (const float*)d_in[7];
  const float* b2    = (const float*)d_in[8];
  const float* gamma = (const float*)d_in[9];
  const float* beta  = (const float*)d_in[10];
  float* out = (float*)d_out;

  const int N = in_sizes[0] / D;
  const int E = in_sizes[3];

  float* ws = (float*)d_ws;
  const size_t nd = (size_t)N * D;
  float* agg_sum = ws;            // [N, D]
  float* agg_max = ws + nd;       // [N, D]
  float* agg_ew  = ws + 2 * nd;   // [N, D]
  float* h2      = ws + 3 * nd;   // [N, D]
  float* deg     = ws + 4 * nd;   // [N]
  float* absum   = deg + N;       // [N]
  float* wsum    = absum + N;     // [N]
  float* stats   = wsum + N;      // [2*D] sums / sumsq

  int initBlocks = (int)((nd + 255) / 256);
  if (initBlocks > 4096) initBlocks = 4096;
  hipLaunchKernelGGL(dgn_init_kernel, dim3(initBlocks), dim3(256), 0, stream,
                     agg_sum, agg_max, agg_ew, deg, absum, wsum, stats, N);
  hipLaunchKernelGGL(dgn_pre_kernel, dim3((E + 255) / 256), dim3(256), 0, stream,
                     eig, dst, deg, absum, E);
  hipLaunchKernelGGL(dgn_edge_kernel, dim3((E + 127) / 128), dim3(256), 0, stream,
                     h, eig, src, dst, W1, b1, absum,
                     agg_sum, agg_max, agg_ew, wsum, E);
  hipLaunchKernelGGL(dgn_node_kernel, dim3((N + 63) / 64), dim3(128), 0, stream,
                     h, snorm, W2, b2, agg_sum, agg_max, agg_ew, deg, wsum,
                     h2, stats, N);
  hipLaunchKernelGGL(dgn_final_kernel, dim3((N * D + 255) / 256), dim3(256), 0, stream,
                     h, h2, stats, gamma, beta, out, N);
}
